// QuantQwenAttention_75505525064736
// MI455X (gfx1250) — 8-bit matrix operand order repaired and checked by reading, compiled, not run on hardware
//
#include <hip/hip_runtime.h>
#include <stdint.h>


typedef __attribute__((ext_vector_type(8))) int v8i;

#define T_SEQ   2048
#define D_MODEL 2048
#define KVDIM   512
#define NH      16
#define NKV     4
#define HD      128

__device__ __forceinline__ int a_kbase(int v, int half) {
    return ((v >> 1) << 4) + half * 8 + (v & 1) * 4;
}
__device__ __forceinline__ int b_kbase(int v, int half) {
    return ((v >> 1) << 4) + half * 8 + (v & 1) * 4;
}

__device__ __forceinline__ float dq_scale(const float* amax) {
    return fmaxf(amax[0], 1e-8f) * (1.0f / 127.0f);
}

__device__ __forceinline__ void async_copy_b128(unsigned lds_off, const void* gaddr) {
    asm volatile("global_load_async_to_lds_b128 %0, %1, off"
                 :
                 : "v"(lds_off), "v"(gaddr)
                 : "memory");
}
__device__ __forceinline__ void wait_asynccnt0() {
    asm volatile("s_wait_asynccnt 0x0" ::: "memory");
}

__global__ __launch_bounds__(256) void amax_kernel(const float* __restrict__ x,
                                                   long long n, float* __restrict__ out) {
    __shared__ float red[256];
    float m = 0.0f;
    for (long long i = (long long)blockIdx.x * blockDim.x + threadIdx.x; i < n;
         i += (long long)gridDim.x * blockDim.x)
        m = fmaxf(m, fabsf(x[i]));
    red[threadIdx.x] = m;
    __syncthreads();
    for (int s = 128; s > 0; s >>= 1) {
        if ((int)threadIdx.x < s) red[threadIdx.x] = fmaxf(red[threadIdx.x], red[threadIdx.x + s]);
        __syncthreads();
    }
    if (threadIdx.x == 0)
        atomicMax((unsigned int*)out, __float_as_uint(red[0]));
}

__global__ __launch_bounds__(256) void quant_kernel(const float* __restrict__ x, long long n,
                                                    const float* __restrict__ amax,
                                                    signed char* __restrict__ q) {
    const float inv = 1.0f / dq_scale(amax);
    for (long long i = (long long)blockIdx.x * blockDim.x + threadIdx.x; i < n;
         i += (long long)gridDim.x * blockDim.x) {
        float v = rintf(x[i] * inv);
        v = fminf(fmaxf(v, -128.0f), 127.0f);
        q[i] = (signed char)(int)v;
    }
}

#define LDSA_STR 80
#define LDSB_STR 80
__global__ __launch_bounds__(256) void gemm_i8_kernel(
    const signed char* __restrict__ A, const signed char* __restrict__ B,
    const float* __restrict__ amaxA, const float* __restrict__ amaxB,
    const float* __restrict__ bias, float* __restrict__ C, int M, int N, int K) {
    __shared__ __align__(16) signed char As[2][128 * LDSA_STR];
    __shared__ __align__(16) signed char Bs[2][64 * LDSB_STR];
    const int tid = threadIdx.x;
    const int wave = tid >> 5, lane = tid & 31;
    const int half = lane >> 4, l16 = lane & 15;
    const int m0 = blockIdx.y * 128;
    const int n0 = blockIdx.x * 64;

    const int ar0 = tid >> 2, ac0 = tid & 3;
    const int ar1 = (tid + 256) >> 2, ac1 = tid & 3;
    const int br = tid >> 2, bc = tid & 3;

    auto stage = [&](int buf, int k0) {
        async_copy_b128((unsigned)(uintptr_t)&As[buf][ar0 * LDSA_STR + ac0 * 16],
                        &A[(long long)(m0 + ar0) * K + k0 + ac0 * 16]);
        async_copy_b128((unsigned)(uintptr_t)&As[buf][ar1 * LDSA_STR + ac1 * 16],
                        &A[(long long)(m0 + ar1) * K + k0 + ac1 * 16]);
        async_copy_b128((unsigned)(uintptr_t)&Bs[buf][br * LDSB_STR + bc * 16],
                        &B[(long long)(n0 + br) * K + k0 + bc * 16]);
    };

    v8i acc[4];
    acc[0] = {}; acc[1] = {}; acc[2] = {}; acc[3] = {};

    const int nsteps = K >> 6;
    stage(0, 0);
    for (int step = 0; step < nsteps; ++step) {
        const int cur = step & 1;
        wait_asynccnt0();
        __syncthreads();
        if (step + 1 < nsteps) stage(1 - cur, (step + 1) * 64);

        const signed char* Ab = As[cur];
        const signed char* Bb = Bs[cur];
        v8i af;
#pragma unroll
        for (int v = 0; v < 8; ++v)
            af[v] = *(const int*)&Ab[(wave * 16 + l16) * LDSA_STR + a_kbase(v, half)];
        v8i bf0, bf1, bf2, bf3;
#pragma unroll
        for (int v = 0; v < 8; ++v) {
            bf0[v] = *(const int*)&Bb[(0 * 16 + l16) * LDSB_STR + b_kbase(v, half)];
            bf1[v] = *(const int*)&Bb[(1 * 16 + l16) * LDSB_STR + b_kbase(v, half)];
            bf2[v] = *(const int*)&Bb[(2 * 16 + l16) * LDSB_STR + b_kbase(v, half)];
            bf3[v] = *(const int*)&Bb[(3 * 16 + l16) * LDSB_STR + b_kbase(v, half)];
        }
        acc[0] = __builtin_amdgcn_wmma_i32_16x16x64_iu8(true, af, true, bf0, acc[0], false, false);
        acc[1] = __builtin_amdgcn_wmma_i32_16x16x64_iu8(true, af, true, bf1, acc[1], false, false);
        acc[2] = __builtin_amdgcn_wmma_i32_16x16x64_iu8(true, af, true, bf2, acc[2], false, false);
        acc[3] = __builtin_amdgcn_wmma_i32_16x16x64_iu8(true, af, true, bf3, acc[3], false, false);
    }

    const float s = dq_scale(amaxA) * dq_scale(amaxB);
#pragma unroll
    for (int j = 0; j < 4; ++j) {
#pragma unroll
        for (int r = 0; r < 8; ++r) {
            int row = m0 + wave * 16 + r + 8 * half;
            int col = n0 + j * 16 + l16;
            v8i a = (j == 0) ? acc[0] : (j == 1) ? acc[1] : (j == 2) ? acc[2] : acc[3];
            float v = (float)a[r] * s + (bias ? bias[col] : 0.0f);
            C[(long long)row * N + col] = v;
        }
    }
}

__global__ __launch_bounds__(256) void rope_kernel(float* __restrict__ x, int nh) {
    int idx = blockIdx.x * blockDim.x + threadIdx.x;
    int total = T_SEQ * nh * 64;
    if (idx >= total) return;
    int ppt = nh * 64;
    int t = idx / ppt;
    int p = idx - t * ppt;
    int h = p >> 6, i = p & 63;
    long long o = (long long)t * (nh * HD) + h * HD + i;
    float angle = (float)t * __expf(-(float)i * 0.14391157f);
    float c = cosf(angle), sn = sinf(angle);
    float a0 = x[o], b0 = x[o + 64];
    x[o]      = a0 * c - b0 * sn;
    x[o + 64] = b0 * c + a0 * sn;
}

__global__ __launch_bounds__(256) void transpose_i8_kernel(const signed char* __restrict__ in,
                                                           signed char* __restrict__ out) {
    int idx = blockIdx.x * blockDim.x + threadIdx.x;
    if (idx >= T_SEQ * KVDIM) return;
    int r = idx / KVDIM, c = idx - r * KVDIM;
    out[(long long)c * T_SEQ + r] = in[idx];
}

#define SS_STR 2052
#define SP_STR 2064
__global__ __launch_bounds__(256) void attn_kernel(
    const signed char* __restrict__ qq, const signed char* __restrict__ kq,
    const signed char* __restrict__ vqT, const float* __restrict__ mask,
    const float* __restrict__ amax_q, const float* __restrict__ amax_k,
    const float* __restrict__ amax_v, float* __restrict__ o) {
    __shared__ float sS[16 * SS_STR];
    __shared__ signed char sP[16 * SP_STR];

    const int h = blockIdx.y;
    const int t0 = blockIdx.x * 16;
    const int kvh = h >> 2;
    const int tid = threadIdx.x;
    const int wave = tid >> 5, lane = tid & 31;
    const int half = lane >> 4, l16 = lane & 15;

    v8i aq0, aq1;
    const signed char* qbase = qq + (long long)(t0 + l16) * D_MODEL + h * HD;
#pragma unroll
    for (int v = 0; v < 8; ++v) {
        aq0[v] = *(const int*)(qbase + a_kbase(v, half));
        aq1[v] = *(const int*)(qbase + 64 + a_kbase(v, half));
    }
    const float sscale = dq_scale(amax_q) * dq_scale(amax_k) * 0.08838834764831845f;

    for (int st = wave; st < T_SEQ / 16; st += 8) {
        int s0 = st * 16;
        v8i bf0, bf1;
        const signed char* kbase = kq + (long long)(s0 + l16) * KVDIM + kvh * HD;
#pragma unroll
        for (int v = 0; v < 8; ++v) {
            bf0[v] = *(const int*)(kbase + b_kbase(v, half));
            bf1[v] = *(const int*)(kbase + 64 + b_kbase(v, half));
        }
        v8i acc = {};
        acc = __builtin_amdgcn_wmma_i32_16x16x64_iu8(true, aq0, true, bf0, acc, false, false);
        acc = __builtin_amdgcn_wmma_i32_16x16x64_iu8(true, aq1, true, bf1, acc, false, false);
#pragma unroll
        for (int r = 0; r < 8; ++r) {
            int tl = r + 8 * half;
            int s = s0 + l16;
            sS[tl * SS_STR + s] =
                (float)acc[r] * sscale + mask[(long long)(t0 + tl) * T_SEQ + s];
        }
    }
    __syncthreads();

    {
        int row = wave * 2 + half;
        float m = -1e30f;
        for (int s = l16; s < T_SEQ; s += 16) m = fmaxf(m, sS[row * SS_STR + s]);
#pragma unroll
        for (int d = 8; d >= 1; d >>= 1) m = fmaxf(m, __shfl_xor(m, d, 16));
        float sum = 0.0f;
        for (int s = l16; s < T_SEQ; s += 16) {
            float e = __expf(sS[row * SS_STR + s] - m);
            sS[row * SS_STR + s] = e;
            sum += e;
        }
#pragma unroll
        for (int d = 8; d >= 1; d >>= 1) sum += __shfl_xor(sum, d, 16);
        float inv = 127.0f / sum;
        for (int s = l16; s < T_SEQ; s += 16) {
            float p = rintf(sS[row * SS_STR + s] * inv);
            sP[row * SP_STR + s] = (signed char)(int)fminf(p, 127.0f);
        }
    }
    __syncthreads();

    const float vscale = dq_scale(amax_v) * (1.0f / 127.0f);
    const int n0 = wave * 16;
    v8i acc = {};
    for (int k0 = 0; k0 < T_SEQ; k0 += 64) {
        v8i af, bf;
#pragma unroll
        for (int v = 0; v < 8; ++v)
            af[v] = *(const int*)&sP[l16 * SP_STR + k0 + a_kbase(v, half)];
        const signed char* vb = vqT + (long long)(kvh * HD + n0 + l16) * T_SEQ + k0;
#pragma unroll
        for (int v = 0; v < 8; ++v)
            bf[v] = *(const int*)(vb + b_kbase(v, half));
        acc = __builtin_amdgcn_wmma_i32_16x16x64_iu8(true, af, true, bf, acc, false, false);
    }
#pragma unroll
    for (int r = 0; r < 8; ++r) {
        int tl = r + 8 * half;
        o[(long long)(t0 + tl) * D_MODEL + h * HD + n0 + l16] = (float)acc[r] * vscale;
    }
}

extern "C" void kernel_launch(void* const* d_in, const int* in_sizes, int n_in,
                              void* d_out, int out_size, void* d_ws, size_t ws_size,
                              hipStream_t stream) {
    (void)in_sizes; (void)n_in; (void)out_size; (void)ws_size;
    const float* hs   = (const float*)d_in[0];
    const float* mask = (const float*)d_in[1];
    const float* Wq = (const float*)d_in[2];
    const float* bq = (const float*)d_in[3];
    const float* Wk = (const float*)d_in[4];
    const float* bk = (const float*)d_in[5];
    const float* Wv = (const float*)d_in[6];
    const float* bv = (const float*)d_in[7];
    const float* Wo = (const float*)d_in[8];
    float* out = (float*)d_out;

    char* ws = (char*)d_ws;
    size_t off = 0;
    auto alloc = [&](size_t bytes) -> void* {
        void* p = ws + off;
        off += (bytes + 255) & ~(size_t)255;
        return p;
    };
    float* amax = (float*)alloc(256);
    signed char* xq  = (signed char*)alloc((size_t)T_SEQ * D_MODEL);
    signed char* wqq = (signed char*)alloc((size_t)D_MODEL * D_MODEL);
    signed char* wkq = (signed char*)alloc((size_t)KVDIM * D_MODEL);
    signed char* wvq = (signed char*)alloc((size_t)KVDIM * D_MODEL);
    signed char* woq = (signed char*)alloc((size_t)D_MODEL * D_MODEL);
    float* qf = (float*)alloc((size_t)T_SEQ * D_MODEL * 4);
    float* kf = (float*)alloc((size_t)T_SEQ * KVDIM * 4);
    float* vf = (float*)alloc((size_t)T_SEQ * KVDIM * 4);
    signed char* qq8 = (signed char*)alloc((size_t)T_SEQ * D_MODEL);
    signed char* kq8 = (signed char*)alloc((size_t)T_SEQ * KVDIM);
    signed char* vq8 = (signed char*)alloc((size_t)T_SEQ * KVDIM);
    signed char* vqT = (signed char*)alloc((size_t)T_SEQ * KVDIM);
    float* of = (float*)alloc((size_t)T_SEQ * D_MODEL * 4);
    signed char* oq8 = (signed char*)alloc((size_t)T_SEQ * D_MODEL);

    hipMemsetAsync(amax, 0, 256, stream);

    const long long nXD = (long long)T_SEQ * D_MODEL;
    const long long nKD = (long long)KVDIM * D_MODEL;
    const long long nTK = (long long)T_SEQ * KVDIM;

    amax_kernel<<<1024, 256, 0, stream>>>(hs, nXD, amax + 0);
    amax_kernel<<<1024, 256, 0, stream>>>(Wq, nXD, amax + 1);
    amax_kernel<<<512, 256, 0, stream>>>(Wk, nKD, amax + 2);
    amax_kernel<<<512, 256, 0, stream>>>(Wv, nKD, amax + 3);
    amax_kernel<<<1024, 256, 0, stream>>>(Wo, nXD, amax + 4);
    quant_kernel<<<2048, 256, 0, stream>>>(hs, nXD, amax + 0, xq);
    quant_kernel<<<2048, 256, 0, stream>>>(Wq, nXD, amax + 1, wqq);
    quant_kernel<<<1024, 256, 0, stream>>>(Wk, nKD, amax + 2, wkq);
    quant_kernel<<<1024, 256, 0, stream>>>(Wv, nKD, amax + 3, wvq);
    quant_kernel<<<2048, 256, 0, stream>>>(Wo, nXD, amax + 4, woq);

    gemm_i8_kernel<<<dim3(D_MODEL / 64, T_SEQ / 128), 256, 0, stream>>>(
        xq, wqq, amax + 0, amax + 1, bq, qf, T_SEQ, D_MODEL, D_MODEL);
    gemm_i8_kernel<<<dim3(KVDIM / 64, T_SEQ / 128), 256, 0, stream>>>(
        xq, wkq, amax + 0, amax + 2, bk, kf, T_SEQ, KVDIM, D_MODEL);
    gemm_i8_kernel<<<dim3(KVDIM / 64, T_SEQ / 128), 256, 0, stream>>>(
        xq, wvq, amax + 0, amax + 3, bv, vf, T_SEQ, KVDIM, D_MODEL);

    rope_kernel<<<(T_SEQ * NH * 64) / 256, 256, 0, stream>>>(qf, NH);
    rope_kernel<<<(T_SEQ * NKV * 64) / 256, 256, 0, stream>>>(kf, NKV);

    amax_kernel<<<1024, 256, 0, stream>>>(qf, nXD, amax + 5);
    amax_kernel<<<512, 256, 0, stream>>>(kf, nTK, amax + 6);
    amax_kernel<<<512, 256, 0, stream>>>(vf, nTK, amax + 7);
    quant_kernel<<<2048, 256, 0, stream>>>(qf, nXD, amax + 5, qq8);
    quant_kernel<<<1024, 256, 0, stream>>>(kf, nTK, amax + 6, kq8);
    quant_kernel<<<1024, 256, 0, stream>>>(vf, nTK, amax + 7, vq8);
    transpose_i8_kernel<<<(T_SEQ * KVDIM) / 256, 256, 0, stream>>>(vq8, vqT);

    attn_kernel<<<dim3(T_SEQ / 16, NH), 256, 0, stream>>>(
        qq8, kq8, vqT, mask, amax + 5, amax + 6, amax + 7, of);

    amax_kernel<<<1024, 256, 0, stream>>>(of, nXD, amax + 8);
    quant_kernel<<<2048, 256, 0, stream>>>(of, nXD, amax + 8, oq8);
    gemm_i8_kernel<<<dim3(D_MODEL / 64, T_SEQ / 128), 256, 0, stream>>>(
        oq8, woq, amax + 8, amax + 4, nullptr, out, T_SEQ, D_MODEL, D_MODEL);
}
